// IGCL_76115410419855
// MI455X (gfx1250) — matrix operand order repaired and checked by reading, not compiled, not run on hardware
//
#include <hip/hip_runtime.h>

#define N_ROWS 100000
#define M_CL   2048
#define E_EDGE 65536
#define IN_DIM 512
#define HID    256
#define TAU    0.5f
#define LAMDA  1.0f

typedef __attribute__((ext_vector_type(16))) __bf16 v16bf;
typedef __attribute__((ext_vector_type(2)))  __bf16 v2bf;
typedef __attribute__((ext_vector_type(8)))  float  v8f;

__device__ __forceinline__ v8f wmma_bf16(v16bf a, v16bf b, v8f c) {
    return __builtin_amdgcn_wmma_f32_16x16x32_bf16(
        false, a, false, b, (short)0, c, false, false);
}

// A-fragment (16x32 bf16) from a [16][lda] tile (LDS or global row base).
// Lane l: row M = l&15; K halves per ISA 7.12.2: lanes<16 hold K {0..7,16..23},
// lanes>=16 hold K {8..15,24..31}.
__device__ __forceinline__ v16bf load_a_frag(const __bf16* __restrict__ A,
                                             int lda, int k0, int lane) {
    const __bf16* row = A + (long)(lane & 15) * lda;
    const int hi  = lane >> 4;
    const int kb0 = k0 + hi * 8;
    const int kb1 = k0 + 16 + hi * 8;
    v16bf a;
#pragma unroll
    for (int i = 0; i < 8; ++i) { a[i] = row[kb0 + i]; a[8 + i] = row[kb1 + i]; }
    return a;
}

// B-fragment (32x16 bf16) from row-major [K][ldb]: lane l holds column N n0+(l&15),
// K halves as A: lanes<16 hold K {0..7,16..23}, lanes>=16 hold K {8..15,24..31}.
__device__ __forceinline__ v16bf load_b_frag(const __bf16* __restrict__ B,
                                             int ldb, int k0, int n0, int lane) {
    const __bf16* col = B + (long)k0 * ldb + n0 + (lane & 15);
    const int hi = lane >> 4;
    v16bf b;
#pragma unroll
    for (int i = 0; i < 8; ++i) { b[i] = col[(long)(hi * 8 + i) * ldb]; b[8 + i] = col[(long)(16 + hi * 8 + i) * ldb]; }
    return b;
}

__global__ void zero_kernel(float* __restrict__ p, long n) {
    long i = (long)blockIdx.x * blockDim.x + threadIdx.x;
    if (i < n) p[i] = 0.0f;
}

__global__ void cvt_bf16_kernel(const float* __restrict__ src,
                                __bf16* __restrict__ dst, int n) {
    int i = blockIdx.x * blockDim.x + threadIdx.x;
    if (i < n) dst[i] = (__bf16)src[i];
}

// ---------------------------------------------------------------------------
// Kernel A: fused encoder + target encoder over ONE read of x.
// Block = 128 threads (4 waves); 32 rows of x per block; wave w owns 64 cols,
// two 16-row M tiles -> 2 WMMAs per B-fragment fetch.
// ---------------------------------------------------------------------------
__global__ __launch_bounds__(128) void enc_fused_kernel(
    const float* __restrict__ x,
    const __bf16* __restrict__ We0b, const float* __restrict__ be0,
    const __bf16* __restrict__ We1b, const float* __restrict__ be1,
    const __bf16* __restrict__ Wt0b, const float* __restrict__ bt0,
    const __bf16* __restrict__ Wt1b, const float* __restrict__ bt1,
    const float* __restrict__ p_vals, const int* __restrict__ p_assign,
    float* __restrict__ hc, __bf16* __restrict__ vn_bf)
{
    __shared__ __bf16 xs[32][IN_DIM];   // 32 KB
    __shared__ __bf16 h1s[32][HID];     // 16 KB
    __shared__ float  vs[32][HID];      // 32 KB
    __shared__ float  psum[32][4];
    __shared__ float  rinv[32];

    const int  tid  = threadIdx.x;
    const int  lane = tid & 31;
    const int  wave = tid >> 5;
    const long r0   = (long)blockIdx.x * 32;
    const int  nbase = wave * 64;

    // stage x tile as bf16 via b128 loads (4 floats / load)
    {
        const float4* x4 = (const float4*)x;
        for (int i = tid; i < 32 * (IN_DIM / 4); i += 128) {
            int r = i >> 7, c4 = i & 127;
            float4 v = x4[(r0 + r) * (IN_DIM / 4) + c4];
            int c = c4 * 4;
            xs[r][c + 0] = (__bf16)v.x;
            xs[r][c + 1] = (__bf16)v.y;
            xs[r][c + 2] = (__bf16)v.z;
            xs[r][c + 3] = (__bf16)v.w;
        }
    }
    __syncthreads();

    for (int br = 0; br < 2; ++br) {
        const __bf16* W0 = br ? Wt0b : We0b;
        const __bf16* W1 = br ? Wt1b : We1b;
        const float*  b0 = br ? bt0 : be0;
        const float*  b1 = br ? bt1 : be1;

        // layer 1: [32,512] @ [512,256] -> [32,256]
        v8f acc[2][4] = {};
        for (int k = 0; k < IN_DIM; k += 32) {
            v16bf a0 = load_a_frag(&xs[0][0],  IN_DIM, k, lane);
            v16bf a1 = load_a_frag(&xs[16][0], IN_DIM, k, lane);
#pragma unroll
            for (int t = 0; t < 4; ++t) {
                v16bf b = load_b_frag(W0, HID, k, nbase + t * 16, lane);
                acc[0][t] = wmma_bf16(a0, b, acc[0][t]);
                acc[1][t] = wmma_bf16(a1, b, acc[1][t]);
            }
        }
#pragma unroll
        for (int mt = 0; mt < 2; ++mt) {
#pragma unroll
            for (int t = 0; t < 4; ++t) {
                int n = nbase + t * 16 + (lane & 15);
                float bias = b0[n];
#pragma unroll
                for (int v = 0; v < 8; ++v) {
                    int m = mt * 16 + v + 8 * (lane >> 4);
                    h1s[m][n] = (__bf16)(acc[mt][t][v] + bias);
                }
            }
        }
        __syncthreads();

        // layer 2: [32,256] @ [256,256] -> [32,256]
        v8f acc2[2][4] = {};
        for (int k = 0; k < HID; k += 32) {
            v16bf a0 = load_a_frag(&h1s[0][0],  HID, k, lane);
            v16bf a1 = load_a_frag(&h1s[16][0], HID, k, lane);
#pragma unroll
            for (int t = 0; t < 4; ++t) {
                v16bf b = load_b_frag(W1, HID, k, nbase + t * 16, lane);
                acc2[0][t] = wmma_bf16(a0, b, acc2[0][t]);
                acc2[1][t] = wmma_bf16(a1, b, acc2[1][t]);
            }
        }

        if (br == 0) {
            // encoder branch: hc[p_assign[n]] += p_vals[n] * h[n]
#pragma unroll
            for (int mt = 0; mt < 2; ++mt) {
#pragma unroll
                for (int t = 0; t < 4; ++t) {
                    int n = nbase + t * 16 + (lane & 15);
                    float bias = b1[n];
#pragma unroll
                    for (int v = 0; v < 8; ++v) {
                        int  m   = mt * 16 + v + 8 * (lane >> 4);
                        long row = r0 + m;
                        int  seg = p_assign[row];
                        float hval = acc2[mt][t][v] + bias;
                        unsafeAtomicAdd(&hc[(long)seg * HID + n],
                                        p_vals[row] * hval);
                    }
                }
            }
            __syncthreads();   // protect h1s before next branch rewrites it
        } else {
            // target branch: vn = l2normalize(v), stored bf16
#pragma unroll
            for (int mt = 0; mt < 2; ++mt) {
#pragma unroll
                for (int t = 0; t < 4; ++t) {
                    int n = nbase + t * 16 + (lane & 15);
                    float bias = b1[n];
#pragma unroll
                    for (int v = 0; v < 8; ++v) {
                        int m = mt * 16 + v + 8 * (lane >> 4);
                        vs[m][n] = acc2[mt][t][v] + bias;
                    }
                }
            }
            __syncthreads();
            {   // parallel row-norms: 4 partials per row
                int r = tid >> 2, qt = tid & 3;
                float s = 0.0f;
                for (int c = qt * 64; c < qt * 64 + 64; ++c) {
                    float t2 = vs[r][c]; s += t2 * t2;
                }
                psum[r][qt] = s;
            }
            __syncthreads();
            if (tid < 32) {
                float s = psum[tid][0] + psum[tid][1] + psum[tid][2] + psum[tid][3];
                rinv[tid] = 1.0f / fmaxf(sqrtf(s), 1e-12f);
            }
            __syncthreads();
            for (int i = tid; i < 32 * (HID / 2); i += 128) {
                int r = i >> 7, c = (i & 127) * 2;
                float sc = rinv[r];
                v2bf o; o[0] = (__bf16)(vs[r][c] * sc); o[1] = (__bf16)(vs[r][c + 1] * sc);
                *(v2bf*)(vn_bf + (r0 + r) * HID + c) = o;
            }
        }
    }
}

// ---------------------------------------------------------------------------
// Kernel B: hn = l2n(hc) (bf16 + transposed bf16), q = l2n(relu(hc@Wp0+bp0)@Wp1+bp1)
// ---------------------------------------------------------------------------
__global__ __launch_bounds__(128) void proj_kernel(
    const float* __restrict__ hc,
    const __bf16* __restrict__ Wp0b, const float* __restrict__ bp0,
    const __bf16* __restrict__ Wp1b, const float* __restrict__ bp1,
    float* __restrict__ q, __bf16* __restrict__ hn_bf, __bf16* __restrict__ hnT_bf)
{
    __shared__ __bf16 hcs[16][HID];
    __shared__ __bf16 h1s[16][HID];
    __shared__ float  fs[16][HID];
    __shared__ float  rinv[16];

    const int tid = threadIdx.x, lane = tid & 31, wave = tid >> 5;
    const int r0 = blockIdx.x * 16;
    const int nbase = wave * 64;

    for (int i = tid; i < 16 * HID; i += 128) {
        int r = i >> 8, c = i & (HID - 1);
        float v = hc[(long)(r0 + r) * HID + c];
        hcs[r][c] = (__bf16)v;
        fs[r][c]  = v;
    }
    __syncthreads();
    if (tid < 16) {
        float s = 0.0f;
        for (int c = 0; c < HID; ++c) { float t = fs[tid][c]; s += t * t; }
        rinv[tid] = 1.0f / fmaxf(sqrtf(s), 1e-12f);
    }
    __syncthreads();
    for (int i = tid; i < 16 * HID; i += 128) {
        int r = i >> 8, c = i & (HID - 1);
        __bf16 hv = (__bf16)(fs[r][c] * rinv[r]);
        hn_bf[(long)(r0 + r) * HID + c]   = hv;
        hnT_bf[(long)c * M_CL + (r0 + r)] = hv;
    }
    __syncthreads();

    // projector layer 1 + relu
    v8f acc[4] = {};
    for (int k = 0; k < HID; k += 32) {
        v16bf a = load_a_frag(&hcs[0][0], HID, k, lane);
#pragma unroll
        for (int t = 0; t < 4; ++t) {
            v16bf b = load_b_frag(Wp0b, HID, k, nbase + t * 16, lane);
            acc[t] = wmma_bf16(a, b, acc[t]);
        }
    }
#pragma unroll
    for (int t = 0; t < 4; ++t) {
        int n = nbase + t * 16 + (lane & 15);
        float bias = bp0[n];
#pragma unroll
        for (int v = 0; v < 8; ++v) {
            int m = v + 8 * (lane >> 4);
            h1s[m][n] = (__bf16)fmaxf(acc[t][v] + bias, 0.0f);
        }
    }
    __syncthreads();

    // projector layer 2
    v8f acc2[4] = {};
    for (int k = 0; k < HID; k += 32) {
        v16bf a = load_a_frag(&h1s[0][0], HID, k, lane);
#pragma unroll
        for (int t = 0; t < 4; ++t) {
            v16bf b = load_b_frag(Wp1b, HID, k, nbase + t * 16, lane);
            acc2[t] = wmma_bf16(a, b, acc2[t]);
        }
    }
#pragma unroll
    for (int t = 0; t < 4; ++t) {
        int n = nbase + t * 16 + (lane & 15);
        float bias = bp1[n];
#pragma unroll
        for (int v = 0; v < 8; ++v) {
            int m = v + 8 * (lane >> 4);
            fs[m][n] = acc2[t][v] + bias;
        }
    }
    __syncthreads();
    if (tid < 16) {
        float s = 0.0f;
        for (int c = 0; c < HID; ++c) { float t = fs[tid][c]; s += t * t; }
        rinv[tid] = 1.0f / fmaxf(sqrtf(s), 1e-12f);
    }
    __syncthreads();
    for (int i = tid; i < 16 * HID; i += 128) {
        int r = i >> 8, c = i & (HID - 1);
        q[(long)(r0 + r) * HID + c] = fs[r][c] * rinv[r];
    }
}

// ---------------------------------------------------------------------------
// Kernel C: pos_acc[seg] += p_vals[n] * dot(vn[n], q[seg]) / TAU  (wave/row)
// ---------------------------------------------------------------------------
__global__ __launch_bounds__(128) void pos_kernel(
    const __bf16* __restrict__ vn_bf, const float* __restrict__ q,
    const float* __restrict__ p_vals, const int* __restrict__ p_assign,
    float* __restrict__ pos_acc)
{
    long w = ((long)blockIdx.x * blockDim.x + threadIdx.x) >> 5;
    int lane = threadIdx.x & 31;
    if (w >= N_ROWS) return;
    int seg = p_assign[w];
    const __bf16* vr = vn_bf + w * HID;
    const float*  qr = q + (long)seg * HID;
    float s = 0.0f;
#pragma unroll
    for (int c = lane; c < HID; c += 32) s += (float)vr[c] * qr[c];
#pragma unroll
    for (int off = 16; off > 0; off >>= 1) s += __shfl_xor(s, off, 32);
    if (lane == 0) unsafeAtomicAdd(&pos_acc[seg], p_vals[w] * s * (1.0f / TAU));
}

// ---------------------------------------------------------------------------
// Kernel D: neg = exp(hn @ hn.T / TAU)  [2048x2048], plus row sums S.
// Grid (M/16, M/256), block 128 (4 waves * 64 j-cols).
// ---------------------------------------------------------------------------
__global__ __launch_bounds__(128) void neg_kernel(
    const __bf16* __restrict__ hn_bf, const __bf16* __restrict__ hnT_bf,
    float* __restrict__ neg, float* __restrict__ S)
{
    const int tid = threadIdx.x, lane = tid & 31, wave = tid >> 5;
    const int r0 = blockIdx.x * 16;
    const int jb = blockIdx.y * 256 + wave * 64;

    v8f acc[4] = {};
    for (int k = 0; k < HID; k += 32) {
        v16bf a = load_a_frag(hn_bf + (long)r0 * HID, HID, k, lane);
#pragma unroll
        for (int t = 0; t < 4; ++t) {
            v16bf b = load_b_frag(hnT_bf, M_CL, k, jb + t * 16, lane);
            acc[t] = wmma_bf16(a, b, acc[t]);
        }
    }

    float rs[8];
#pragma unroll
    for (int v = 0; v < 8; ++v) rs[v] = 0.0f;
#pragma unroll
    for (int t = 0; t < 4; ++t) {
        int n = jb + t * 16 + (lane & 15);
#pragma unroll
        for (int v = 0; v < 8; ++v) {
            int m = v + 8 * (lane >> 4);
            float e = __expf(acc[t][v] * (1.0f / TAU));
            neg[(long)(r0 + m) * M_CL + n] = e;
            rs[v] += e;
        }
    }
    // reduce over the 16 lanes of each half (masks < 16 stay within a half)
#pragma unroll
    for (int v = 0; v < 8; ++v) {
        float s = rs[v];
        s += __shfl_xor(s, 1, 32); s += __shfl_xor(s, 2, 32);
        s += __shfl_xor(s, 4, 32); s += __shfl_xor(s, 8, 32);
        if ((lane & 15) == 0) {
            int m = v + 8 * (lane >> 4);
            unsafeAtomicAdd(&S[r0 + m], s);
        }
    }
}

// ---------------------------------------------------------------------------
// Kernel E: ps[cg_rows[e], :] += cg_vals[e] * neg[cg_cols[e], :]
// ---------------------------------------------------------------------------
__global__ __launch_bounds__(256) void scatter_kernel(
    const float* __restrict__ cg_vals, const int* __restrict__ cg_rows,
    const int* __restrict__ cg_cols, const float* __restrict__ neg,
    float* __restrict__ ps)
{
    int e = blockIdx.x;
    float w = cg_vals[e];
    const float* src = neg + (long)cg_cols[e] * M_CL;
    float*       dst = ps  + (long)cg_rows[e] * M_CL;
    for (int j = threadIdx.x; j < M_CL; j += 256)
        unsafeAtomicAdd(&dst[j], w * src[j]);
}

// ---------------------------------------------------------------------------
// Kernel F: loss = mean(-log(pos + L*ps)) + mean_m log(pos + S)
// ---------------------------------------------------------------------------
__global__ __launch_bounds__(256) void loss_kernel(
    const float* __restrict__ pos_acc, const float* __restrict__ S,
    const float* __restrict__ ps, float* __restrict__ out)
{
    const long  total = (long)M_CL * M_CL;
    const float invMM = 1.0f / ((float)M_CL * (float)M_CL);
    float local = 0.0f;
    for (long idx = (long)blockIdx.x * 256 + threadIdx.x; idx < total;
         idx += (long)gridDim.x * 256) {
        int m = (int)(idx >> 11);   // / 2048
        float pm = __expf(pos_acc[m]);
        float t = -__logf(pm + LAMDA * ps[idx]);
        if ((idx & (M_CL - 1)) == 0) t += (float)M_CL * __logf(pm + S[m]);
        local += t;
    }
    local *= invMM;
    __shared__ float red[8];
#pragma unroll
    for (int off = 16; off > 0; off >>= 1) local += __shfl_xor(local, off, 32);
    int lane = threadIdx.x & 31, wave = threadIdx.x >> 5;
    if (lane == 0) red[wave] = local;
    __syncthreads();
    if (threadIdx.x == 0) {
        float s = 0.0f;
        for (int i = 0; i < 8; ++i) s += red[i];
        unsafeAtomicAdd(out, s);
    }
}

// ---------------------------------------------------------------------------
extern "C" void kernel_launch(void* const* d_in, const int* in_sizes, int n_in,
                              void* d_out, int out_size, void* d_ws, size_t ws_size,
                              hipStream_t stream) {
    const float* x      = (const float*)d_in[0];
    const float* We0    = (const float*)d_in[1];
    const float* be0    = (const float*)d_in[2];
    const float* We1    = (const float*)d_in[3];
    const float* be1    = (const float*)d_in[4];
    const float* Wt0    = (const float*)d_in[5];
    const float* bt0    = (const float*)d_in[6];
    const float* Wt1    = (const float*)d_in[7];
    const float* bt1    = (const float*)d_in[8];
    const float* Wp0    = (const float*)d_in[9];
    const float* bp0    = (const float*)d_in[10];
    const float* Wp1    = (const float*)d_in[11];
    const float* bp1    = (const float*)d_in[12];
    const float* p_vals = (const float*)d_in[13];
    const float* cg_vals= (const float*)d_in[14];
    const int*   p_asgn = (const int*)d_in[15];
    const int*   cg_rows= (const int*)d_in[16];
    const int*   cg_cols= (const int*)d_in[17];
    // d_in[18] = neg_idx (unused by the reference computation)

    // ---- workspace layout ----
    char* base = (char*)d_ws;
    size_t off = 0;
    auto carve = [&](size_t bytes) -> char* {
        char* p = base + off;
        off += (bytes + 255) & ~(size_t)255;
        return p;
    };
    __bf16* We0b = (__bf16*)carve((size_t)IN_DIM * HID * 2);
    __bf16* We1b = (__bf16*)carve((size_t)HID * HID * 2);
    __bf16* Wt0b = (__bf16*)carve((size_t)IN_DIM * HID * 2);
    __bf16* Wt1b = (__bf16*)carve((size_t)HID * HID * 2);
    __bf16* Wp0b = (__bf16*)carve((size_t)HID * HID * 2);
    __bf16* Wp1b = (__bf16*)carve((size_t)HID * HID * 2);
    __bf16* vn_bf = (__bf16*)carve((size_t)N_ROWS * HID * 2);
    float*  hc    = (float*)carve((size_t)M_CL * HID * 4);
    float*  q     = (float*)carve((size_t)M_CL * HID * 4);
    __bf16* hn_bf = (__bf16*)carve((size_t)M_CL * HID * 2);
    __bf16* hnT_bf= (__bf16*)carve((size_t)HID * M_CL * 2);
    float*  pos_acc = (float*)carve((size_t)M_CL * 4);
    float*  Ssum    = (float*)carve((size_t)M_CL * 4);
    float*  neg     = (float*)carve((size_t)M_CL * M_CL * 4);
    float*  ps      = (float*)carve((size_t)M_CL * M_CL * 4);

    // ---- zero accumulators + output ----
    auto zero = [&](float* p, long n) {
        int blocks = (int)((n + 255) / 256);
        zero_kernel<<<blocks, 256, 0, stream>>>(p, n);
    };
    zero(hc, (long)M_CL * HID);
    zero(pos_acc, M_CL);
    zero(Ssum, M_CL);
    zero(ps, (long)M_CL * M_CL);
    zero((float*)d_out, out_size);

    // ---- convert weights to bf16 ----
    auto cvt = [&](const float* s, __bf16* d, int n) {
        cvt_bf16_kernel<<<(n + 255) / 256, 256, 0, stream>>>(s, d, n);
    };
    cvt(We0, We0b, IN_DIM * HID);
    cvt(We1, We1b, HID * HID);
    cvt(Wt0, Wt0b, IN_DIM * HID);
    cvt(Wt1, Wt1b, HID * HID);
    cvt(Wp0, Wp0b, HID * HID);
    cvt(Wp1, Wp1b, HID * HID);

    // ---- fused encoders (one pass over x, 32 rows/block) ----
    enc_fused_kernel<<<N_ROWS / 32, 128, 0, stream>>>(
        x, We0b, be0, We1b, be1, Wt0b, bt0, Wt1b, bt1,
        p_vals, p_asgn, hc, vn_bf);

    // ---- projector + hn ----
    proj_kernel<<<M_CL / 16, 128, 0, stream>>>(
        hc, Wp0b, bp0, Wp1b, bp1, q, hn_bf, hnT_bf);

    // ---- positive scores ----
    pos_kernel<<<(N_ROWS * 32 + 127) / 128, 128, 0, stream>>>(
        vn_bf, q, p_vals, p_asgn, pos_acc);

    // ---- negative score GEMM + row sums ----
    dim3 ngrid(M_CL / 16, M_CL / 256);
    neg_kernel<<<ngrid, 128, 0, stream>>>(hn_bf, hnT_bf, neg, Ssum);

    // ---- sparse partition scatter ----
    scatter_kernel<<<E_EDGE, 256, 0, stream>>>(cg_vals, cg_rows, cg_cols, neg, ps);

    // ---- loss reduction ----
    loss_kernel<<<2048, 256, 0, stream>>>(pos_acc, Ssum, ps, (float*)d_out);
}
